// PerResidualScaledEncoderBlock_21998822490699
// MI455X (gfx1250) — compile-verified, matrix operand order checked by reading, not run on hardware
//
#include <hip/hip_runtime.h>
#include <hip/hip_bf16.h>
#include <math.h>

// ---------------------------------------------------------------------------
// Fixed problem shape (from the reference): B=2, S=2048, E=1024, H=16, HD=64,
// FF=4096.  All tiles divide evenly -> no bounds checks, EXEC always all-1s.
// ---------------------------------------------------------------------------
constexpr int Bc  = 2;
constexpr int Sc  = 2048;
constexpr int Ec  = 1024;
constexpr int Hc  = 16;
constexpr int HDc = 64;
constexpr int FFc = 4096;
constexpr int MT  = Bc * Sc;          // 4096 token rows

typedef __attribute__((ext_vector_type(16))) __bf16 v16bf;
typedef __attribute__((ext_vector_type(8)))  __bf16 v8bf;
typedef __attribute__((ext_vector_type(8)))  float  v8f;

#define USE_ASYNC_LDS 1

static __device__ __forceinline__ v16bf cat8(v8bf lo, v8bf hi) {
  return __builtin_shufflevector(lo, hi, 0,1,2,3,4,5,6,7,8,9,10,11,12,13,14,15);
}
static __device__ __forceinline__ v8f wmma_bf16(v16bf a, v16bf b, v8f c) {
  // v_wmma_f32_16x16x32_bf16  (8 args: neg_a, A, neg_b, B, c_mod, C, reuse_a, reuse_b)
  return __builtin_amdgcn_wmma_f32_16x16x32_bf16(false, a, false, b, (short)0, c,
                                                 false, false);
}
static __device__ __forceinline__ float gelu_exact(float x) {
  return 0.5f * x * (1.0f + erff(x * 0.70710678118654752f));
}

#if USE_ASYNC_LDS
// Async global->LDS copy, 16B per lane, tracked with ASYNCcnt.  The LDS
// destination address is the low 32 bits of the generic pointer (wave-relative
// LDS byte offset).  ISA: GLOBAL_LOAD_ASYNC_TO_LDS_B128 (VGLOBAL op 98).
static __device__ __forceinline__ void async_ld_b128(void* lds_ptr,
                                                     const void* gptr) {
  const unsigned loff = (unsigned)(unsigned long long)lds_ptr;
  asm volatile("global_load_async_to_lds_b128 %0, %1, off"
               :: "v"(loff), "v"(gptr) : "memory");
}
static __device__ __forceinline__ void wait_async0() {
  asm volatile("s_wait_asynccnt 0" ::: "memory");
}
#endif

// ---------------------------------------------------------------------------
// Weight transform: fp32 [rows, cols] -> bf16 transposed [cols, rows].
// Makes every GEMM "NT": both WMMA fragments become contiguous-K b128 loads.
// ---------------------------------------------------------------------------
__global__ __launch_bounds__(256) void transpose_cvt_kernel(
    const float* __restrict__ W, __bf16* __restrict__ Wt, int rows, int cols) {
  __shared__ float tile[32][33];
  int c0 = blockIdx.x * 32, r0 = blockIdx.y * 32;
  int tx = threadIdx.x, ty = threadIdx.y;          // 32 x 8
#pragma unroll
  for (int i = 0; i < 32; i += 8)
    tile[ty + i][tx] = W[(size_t)(r0 + ty + i) * cols + (c0 + tx)];
  __syncthreads();
#pragma unroll
  for (int i = 0; i < 32; i += 8)
    Wt[(size_t)(c0 + ty + i) * rows + (r0 + tx)] = (__bf16)tile[tx][ty + i];
}

// ---------------------------------------------------------------------------
// LayerNorm: one 256-thread block per token row, emits bf16 GEMM input.
// ---------------------------------------------------------------------------
__global__ __launch_bounds__(256) void ln_bf16_kernel(
    const float* __restrict__ x, const float* __restrict__ g,
    const float* __restrict__ b, __bf16* __restrict__ y) {
  __shared__ float s_sum[256];
  __shared__ float s_sq[256];
  const int row = blockIdx.x, t = threadIdx.x;
  const float4 v = reinterpret_cast<const float4*>(x + (size_t)row * Ec)[t];
  float s  = v.x + v.y + v.z + v.w;
  float sq = v.x * v.x + v.y * v.y + v.z * v.z + v.w * v.w;
  s_sum[t] = s; s_sq[t] = sq;
  __syncthreads();
#pragma unroll
  for (int off = 128; off > 0; off >>= 1) {
    if (t < off) { s_sum[t] += s_sum[t + off]; s_sq[t] += s_sq[t + off]; }
    __syncthreads();
  }
  const float mean = s_sum[0] * (1.0f / Ec);
  const float var  = s_sq[0] * (1.0f / Ec) - mean * mean;
  const float rstd = rsqrtf(var + 1e-5f);
  const int i0 = t * 4;
  const float xi[4] = {v.x, v.y, v.z, v.w};
#pragma unroll
  for (int j = 0; j < 4; ++j)
    y[(size_t)row * Ec + i0 + j] =
        (__bf16)((xi[j] - mean) * rstd * g[i0 + j] + b[i0 + j]);
}

// ---------------------------------------------------------------------------
// NT GEMM: C[M,N] = A[M,K] * Bt[N,K]^T  (bf16 in, fp32 accumulate via WMMA).
// Block tile 128x128, 8 waves of 32 (wave32), wave tile 32x64 (2x4 C-frags).
// Double-buffered async global->LDS staging (ASYNCcnt): tile k+1 streams into
// the shadow buffer while WMMAs consume tile k; one s_wait_asynccnt 0 +
// one barrier per K-step covers both RAW and WAR hazards.
// LDS staging stride 40 bf16 (80 B): conflict-free, 16B-aligned frag reads.
// EPI: 1 = bias+residual -> fp32 out (proj / fc1)
//      2 = bias+GELU     -> bf16 out (fc0)
//      3 = bias + QKV head scatter (Q,K row-major, V transposed) -> bf16
// ---------------------------------------------------------------------------
constexpr int BM = 128, BN = 128, BK = 32, LDT = 40;

template <int EPI>
__global__ __launch_bounds__(256) void gemm_nt_kernel(
    const __bf16* __restrict__ A, const __bf16* __restrict__ Bt,
    const float* __restrict__ bias,
    float* __restrict__ outF, __bf16* __restrict__ outB,
    const float* __restrict__ resid,
    __bf16* __restrict__ qP, __bf16* __restrict__ kP, __bf16* __restrict__ vtP,
    int M, int N, int K) {
  __shared__ alignas(16) __bf16 As[2][BM * LDT];
  __shared__ alignas(16) __bf16 Bs[2][BN * LDT];

  const int t = threadIdx.x;
  const int wid = t >> 5, lane = t & 31, lrow = lane & 15, half = lane >> 4;
  const int wm = wid >> 1, wn = wid & 1;              // 4 x 2 wave grid
  const int m0 = blockIdx.y * BM, n0 = blockIdx.x * BN;

  const v8f vzero = {0.f, 0.f, 0.f, 0.f, 0.f, 0.f, 0.f, 0.f};
  v8f acc[2][4];
#pragma unroll
  for (int mi = 0; mi < 2; ++mi)
#pragma unroll
    for (int ni = 0; ni < 4; ++ni) acc[mi][ni] = vzero;

  const int arow = t >> 1, aseg = (t & 1) * 16;       // 128 rows x 2 segs of 16

  // ---- staging: each thread moves 2x16B of A and 2x16B of B per tile ----
  auto stage = [&](int buf, int k0) {
    const __bf16* ag = A  + (size_t)(m0 + arow) * K + k0 + aseg;
    const __bf16* bg = Bt + (size_t)(n0 + arow) * K + k0 + aseg;
    __bf16* al = &As[buf][arow * LDT + aseg];
    __bf16* bl = &Bs[buf][arow * LDT + aseg];
#if USE_ASYNC_LDS
    async_ld_b128(al,     ag);
    async_ld_b128(al + 8, ag + 8);
    async_ld_b128(bl,     bg);
    async_ld_b128(bl + 8, bg + 8);
#else
    *(v8bf*)al       = *(const v8bf*)ag;
    *(v8bf*)(al + 8) = *(const v8bf*)(ag + 8);
    *(v8bf*)bl       = *(const v8bf*)bg;
    *(v8bf*)(bl + 8) = *(const v8bf*)(bg + 8);
#endif
  };

  stage(0, 0);
#if USE_ASYNC_LDS
  wait_async0();
#endif
  __syncthreads();

  int cur = 0;
  for (int k0 = 0; k0 < K; k0 += BK) {
    if (k0 + BK < K) stage(cur ^ 1, k0 + BK);   // prefetch next tile (async)

    v16bf af[2], bf[4];
#pragma unroll
    for (int mi = 0; mi < 2; ++mi) {
      const int r = wm * 32 + mi * 16 + lrow;
      af[mi] = cat8(*(const v8bf*)&As[cur][r * LDT + half * 8],
                    *(const v8bf*)&As[cur][r * LDT + half * 8 + 16]);
    }
#pragma unroll
    for (int ni = 0; ni < 4; ++ni) {
      const int r = wn * 64 + ni * 16 + lrow;
      bf[ni] = cat8(*(const v8bf*)&Bs[cur][r * LDT + half * 8],
                    *(const v8bf*)&Bs[cur][r * LDT + half * 8 + 16]);
    }
#pragma unroll
    for (int mi = 0; mi < 2; ++mi)
#pragma unroll
      for (int ni = 0; ni < 4; ++ni)
        acc[mi][ni] = wmma_bf16(af[mi], bf[ni], acc[mi][ni]);

#if USE_ASYNC_LDS
    wait_async0();     // my async stores to buf^1 have landed in LDS
#endif
    __syncthreads();   // everyone done reading buf AND everyone's async done
    cur ^= 1;
  }

  // Epilogue.  C-frag element v: m = v + 8*half (lane half), n = lane%16.
#pragma unroll
  for (int mi = 0; mi < 2; ++mi) {
#pragma unroll
    for (int ni = 0; ni < 4; ++ni) {
      const int n  = n0 + wn * 64 + ni * 16 + lrow;
      const int mB = m0 + wm * 32 + mi * 16 + 8 * half;
      const float bv = bias[n];
#pragma unroll
      for (int v = 0; v < 8; ++v) {
        const int m = mB + v;
        float val = acc[mi][ni][v] + bv;
        if constexpr (EPI == 1) {
          val += resid[(size_t)m * N + n];
          outF[(size_t)m * N + n] = val;
        } else if constexpr (EPI == 2) {
          outB[(size_t)m * N + n] = (__bf16)gelu_exact(val);
        } else {  // EPI == 3 : QKV scatter into per-head layouts
          const int bb = m >> 11;               // m / Sc
          const int ss = m & (Sc - 1);
          const int hh = n / (3 * HDc);
          const int jj = n - hh * (3 * HDc);
          const size_t head = (size_t)bb * Hc + hh;
          const __bf16 bv16 = (__bf16)val;
          if (jj < HDc)
            qP[(head * Sc + ss) * HDc + jj] = bv16;
          else if (jj < 2 * HDc)
            kP[(head * Sc + ss) * HDc + (jj - HDc)] = bv16;
          else                                   // V stored transposed [HD,S]
            vtP[(head * HDc + (jj - 2 * HDc)) * Sc + ss] = bv16;
        }
      }
    }
  }
}

// ---------------------------------------------------------------------------
// Flash attention (no 1/sqrt(d) scale, per reference).  Grid (S/64, B*H),
// 4 waves/WG; each wave owns 16 query rows.  QK^T and P*V are both NT WMMAs;
// P goes through a per-wave LDS buffer to convert C-frag -> A-frag layout.
// Next K/V tiles are touched with global_prefetch_b8 to hide stream latency.
// ---------------------------------------------------------------------------
constexpr int PLD = 72;  // P row stride in bf16 (144 B, 16B-aligned chunks)

__global__ __launch_bounds__(128) void attn_kernel(
    const __bf16* __restrict__ Q, const __bf16* __restrict__ K,
    const __bf16* __restrict__ Vt, __bf16* __restrict__ attOut) {
  __shared__ alignas(16) __bf16 Plds[4][16 * PLD];

  const int t = threadIdx.x;
  const int w = t >> 5, lane = t & 31, lrow = lane & 15, half = lane >> 4;
  const int bh = blockIdx.y;
  const int q0 = blockIdx.x * 64 + w * 16;
  const __bf16* qb = Q  + (size_t)bh * Sc * HDc;
  const __bf16* kb = K  + (size_t)bh * Sc * HDc;
  const __bf16* vb = Vt + (size_t)bh * HDc * Sc;
  __bf16* Pl = &Plds[w][0];

  // Q fragments (16 x 64, two K-steps of 32) held for the whole pass.
  v16bf qf[2];
#pragma unroll
  for (int kk = 0; kk < 2; ++kk) {
    const __bf16* p = qb + (size_t)(q0 + lrow) * HDc + kk * 32 + half * 8;
    qf[kk] = cat8(*(const v8bf*)p, *(const v8bf*)(p + 16));
  }

  const v8f vzero = {0.f, 0.f, 0.f, 0.f, 0.f, 0.f, 0.f, 0.f};
  v8f o[4];
#pragma unroll
  for (int dt = 0; dt < 4; ++dt) o[dt] = vzero;
  float mstat[8], lstat[8];
#pragma unroll
  for (int v = 0; v < 8; ++v) { mstat[v] = -1e30f; lstat[v] = 0.f; }

  for (int jt = 0; jt < Sc; jt += 64) {
    // Prefetch next K/V tiles (speculative; dropped if past the buffer).
    __builtin_prefetch(kb + (size_t)(jt + 64 + lane) * HDc, 0, 1);
    __builtin_prefetch(vb + (size_t)(lane)*Sc + jt + 64, 0, 1);

    // S tile = Q * K^T  (16 x 64 as four 16x16 C-frags)
    v8f sf[4];
#pragma unroll
    for (int nt = 0; nt < 4; ++nt) {
      const int j0 = jt + nt * 16;
      v16bf kf0, kf1;
      {
        const __bf16* p = kb + (size_t)(j0 + lrow) * HDc + half * 8;
        kf0 = cat8(*(const v8bf*)p, *(const v8bf*)(p + 16));
        const __bf16* p2 = p + 32;
        kf1 = cat8(*(const v8bf*)p2, *(const v8bf*)(p2 + 16));
      }
      v8f z = vzero;
      z = wmma_bf16(qf[0], kf0, z);
      z = wmma_bf16(qf[1], kf1, z);
      sf[nt] = z;
    }

    // Online softmax.  Row m = v + 8*half lives across a 16-lane group.
#pragma unroll
    for (int v = 0; v < 8; ++v) {
      float mx = fmaxf(fmaxf(sf[0][v], sf[1][v]), fmaxf(sf[2][v], sf[3][v]));
#pragma unroll
      for (int off = 1; off < 16; off <<= 1)
        mx = fmaxf(mx, __shfl_xor(mx, off, 32));
      const float mn = fmaxf(mstat[v], mx);
      const float scale = __expf(mstat[v] - mn);
      mstat[v] = mn;
      float rs = 0.f;
#pragma unroll
      for (int nt = 0; nt < 4; ++nt) {
        const float pexp = __expf(sf[nt][v] - mn);
        sf[nt][v] = pexp;
        rs += pexp;
      }
#pragma unroll
      for (int off = 1; off < 16; off <<= 1) rs += __shfl_xor(rs, off, 32);
      lstat[v] = lstat[v] * scale + rs;
#pragma unroll
      for (int dt = 0; dt < 4; ++dt) o[dt][v] *= scale;
    }

    // C-frag -> LDS (bf16) -> A-frag for P*V.
#pragma unroll
    for (int nt = 0; nt < 4; ++nt)
#pragma unroll
      for (int v = 0; v < 8; ++v)
        Pl[(v + 8 * half) * PLD + nt * 16 + lrow] = (__bf16)sf[nt][v];
    asm volatile("s_wait_dscnt 0" ::: "memory");  // per-wave LDS RAW fence

#pragma unroll
    for (int kk = 0; kk < 2; ++kk) {
      const int jc = kk * 32;
      const v16bf pf =
          cat8(*(const v8bf*)&Pl[lrow * PLD + jc + half * 8],
               *(const v8bf*)&Pl[lrow * PLD + jc + half * 8 + 16]);
#pragma unroll
      for (int dt = 0; dt < 4; ++dt) {
        const __bf16* vp =
            vb + (size_t)(dt * 16 + lrow) * Sc + jt + jc + half * 8;
        const v16bf vf = cat8(*(const v8bf*)vp, *(const v8bf*)(vp + 16));
        o[dt] = wmma_bf16(pf, vf, o[dt]);
      }
    }
  }

  // Write O / l into [B, S, H*HD] bf16 for the proj GEMM.
  const int bb = bh >> 4, hh = bh & 15;
#pragma unroll
  for (int v = 0; v < 8; ++v) {
    const float inv = 1.0f / lstat[v];
    const int ss = q0 + v + 8 * half;
#pragma unroll
    for (int dt = 0; dt < 4; ++dt) {
      const int dd = dt * 16 + lrow;
      attOut[((size_t)bb * Sc + ss) * Ec + hh * HDc + dd] =
          (__bf16)(o[dt][v] * inv);
    }
  }
}

// ---------------------------------------------------------------------------
// Host-side orchestration (graph-capture safe: only kernel launches).
// ---------------------------------------------------------------------------
extern "C" void kernel_launch(void* const* d_in, const int* in_sizes, int n_in,
                              void* d_out, int out_size, void* d_ws,
                              size_t ws_size, hipStream_t stream) {
  const float* x      = (const float*)d_in[0];
  const float* ln1_g  = (const float*)d_in[1];
  const float* ln1_b  = (const float*)d_in[2];
  const float* qkv_w  = (const float*)d_in[3];
  const float* qkv_b  = (const float*)d_in[4];
  const float* proj_w = (const float*)d_in[5];
  const float* proj_b = (const float*)d_in[6];
  const float* ln2_g  = (const float*)d_in[7];
  const float* ln2_b  = (const float*)d_in[8];
  const float* fc0_w  = (const float*)d_in[9];
  const float* fc0_b  = (const float*)d_in[10];
  const float* fc1_w  = (const float*)d_in[11];
  const float* fc1_b  = (const float*)d_in[12];
  float* out = (float*)d_out;

  char* p = (char*)d_ws;
  auto take = [&](size_t bytes) {
    char* r = p;
    p += (bytes + 255) & ~(size_t)255;
    return r;
  };
  __bf16* wqkvT = (__bf16*)take((size_t)3 * Ec * Ec * 2);   // [3E, E]
  __bf16* wprjT = (__bf16*)take((size_t)Ec * Ec * 2);       // [E, E]
  __bf16* wfc0T = (__bf16*)take((size_t)FFc * Ec * 2);      // [FF, E]
  __bf16* wfc1T = (__bf16*)take((size_t)Ec * FFc * 2);      // [E, FF]
  __bf16* ln1o  = (__bf16*)take((size_t)MT * Ec * 2);
  __bf16* qbuf  = (__bf16*)take((size_t)MT * Ec * 2);       // [B,H,S,HD]
  __bf16* kbuf  = (__bf16*)take((size_t)MT * Ec * 2);
  __bf16* vtbuf = (__bf16*)take((size_t)MT * Ec * 2);       // [B,H,HD,S]
  __bf16* attb  = (__bf16*)take((size_t)MT * Ec * 2);       // [B*S, E]
  float*  res2  = (float*)take((size_t)MT * Ec * 4);
  __bf16* ln2o  = (__bf16*)take((size_t)MT * Ec * 2);
  __bf16* hff   = (__bf16*)take((size_t)MT * FFc * 2);      // [B*S, FF]

  const dim3 tb(32, 8);
  transpose_cvt_kernel<<<dim3(96, 32),  tb, 0, stream>>>(qkv_w,  wqkvT, Ec,  3 * Ec);
  transpose_cvt_kernel<<<dim3(32, 32),  tb, 0, stream>>>(proj_w, wprjT, Ec,  Ec);
  transpose_cvt_kernel<<<dim3(128, 32), tb, 0, stream>>>(fc0_w,  wfc0T, Ec,  FFc);
  transpose_cvt_kernel<<<dim3(32, 128), tb, 0, stream>>>(fc1_w,  wfc1T, FFc, Ec);

  ln_bf16_kernel<<<MT, 256, 0, stream>>>(x, ln1_g, ln1_b, ln1o);

  gemm_nt_kernel<3><<<dim3(3 * Ec / BN, MT / BM), 256, 0, stream>>>(
      ln1o, wqkvT, qkv_b, nullptr, nullptr, nullptr, qbuf, kbuf, vtbuf,
      MT, 3 * Ec, Ec);

  attn_kernel<<<dim3(Sc / 64, Bc * Hc), 128, 0, stream>>>(qbuf, kbuf, vtbuf, attb);

  gemm_nt_kernel<1><<<dim3(Ec / BN, MT / BM), 256, 0, stream>>>(
      attb, wprjT, proj_b, res2, nullptr, x, nullptr, nullptr, nullptr,
      MT, Ec, Ec);

  ln_bf16_kernel<<<MT, 256, 0, stream>>>(res2, ln2_g, ln2_b, ln2o);

  gemm_nt_kernel<2><<<dim3(FFc / BN, MT / BM), 256, 0, stream>>>(
      ln2o, wfc0T, fc0_b, nullptr, hff, nullptr, nullptr, nullptr, nullptr,
      MT, FFc, Ec);

  gemm_nt_kernel<1><<<dim3(Ec / BN, MT / BM), 256, 0, stream>>>(
      hff, wfc1T, fc1_b, out, nullptr, res2, nullptr, nullptr, nullptr,
      MT, Ec, FFc);
}
